// CNNLC1_82429012345508
// MI455X (gfx1250) — matrix operand order repaired and checked by reading, not compiled, not run on hardware
//
#include <hip/hip_runtime.h>

typedef __attribute__((ext_vector_type(16))) _Float16 v16h;
typedef __attribute__((ext_vector_type(8)))  _Float16 v8h;
typedef __attribute__((ext_vector_type(8)))  float    v8f;

#define WMMA_F32_F16(A,B,C) \
  __builtin_amdgcn_wmma_f32_16x16x32_f16(false,(A),false,(B),(short)0,(C),false,false)

static __device__ __forceinline__ v16h cat16(v8h lo, v8h hi) {
  return __builtin_shufflevector(lo, hi, 0,1,2,3,4,5,6,7,8,9,10,11,12,13,14,15);
}

// Async global<->LDS DMA (CDNA5, tracked by ASYNCcnt). LDS operand is the
// low 32 bits of the generic pointer.
static __device__ __forceinline__ void g2l_b128(void* lds_dst, const void* gsrc) {
  asm volatile("global_load_async_to_lds_b128 %0, %1, off"
               :: "v"((unsigned)(unsigned long long)lds_dst), "v"(gsrc)
               : "memory");
}
static __device__ __forceinline__ void l2g_b128(void* gdst, const void* lds_src) {
  asm volatile("global_store_async_from_lds_b128 %0, %1, off"
               :: "v"(gdst), "v"((unsigned)(unsigned long long)lds_src)
               : "memory");
}
static __device__ __forceinline__ void wait_async0() {
  asm volatile("s_wait_asynccnt 0" ::: "memory");
}

// ---------------------------------------------------------------------------
// k_prep: convert weights into f16 layouts where every WMMA B-fragment is a
// contiguous [n][k] run (2x16B loads per fragment):
//   wlcT [ij][o(32)][k(32)] : LC weights; k=27 carries the bias (A row 27=1),
//                             k=28..31 zero-padded
//   w1h  [n(256)][k(4096)]  : FC1 weights (straight f16 convert)
//   w2T  [o(64)][k=(dy*3+dx)*32+c] : conv weights, channel-contiguous K
//   w3T  [n(16,pad from 10)][k(256)] : FC2 weights padded
// ---------------------------------------------------------------------------
__global__ __launch_bounds__(256) void k_prep(
    const float* __restrict__ w_lc, const float* __restrict__ b_lc,
    const float* __restrict__ w_conv, const float* __restrict__ w_fc1,
    const float* __restrict__ w_fc2,
    _Float16* __restrict__ wlcT, _Float16* __restrict__ w1h,
    _Float16* __restrict__ w2T, _Float16* __restrict__ w3T)
{
  int g = blockIdx.x * 256 + threadIdx.x;
  const int N1 = 1024 * 32 * 32;   // wlcT
  const int N2 = 256 * 4096;       // w1h
  const int N3 = 64 * 288;         // w2T
  const int N4 = 16 * 256;         // w3T
  if (g < N1) {
    int ij = g >> 10, o = (g >> 5) & 31, k = g & 31;
    float v = 0.f;
    if (k < 27)       v = w_lc[(o * 27 + k) * 1024 + ij];
    else if (k == 27) v = b_lc[o * 1024 + ij];   // bias row (A supplies 1.0)
    wlcT[g] = (_Float16)v;
    return;
  }
  g -= N1;
  if (g < N2) {                     // identical [n][k] layout: pure convert
    w1h[g] = (_Float16)w_fc1[g];
    return;
  }
  g -= N2;
  if (g < N3) {
    int o = g / 288, k = g - o * 288;
    int s = k >> 5, c = k & 31;     // s = dy*3+dx
    w2T[g] = (_Float16)w_conv[o * 288 + c * 9 + s];
    return;
  }
  g -= N3;
  if (g < N4) {
    int n = g >> 8, k = g & 255;
    w3T[g] = (n < 10) ? (_Float16)w_fc2[n * 256 + k] : (_Float16)0.0f;
  }
}

// ---------------------------------------------------------------------------
// k_lc: locally-connected 3x3 (unshared weights) + bias + ReLU + 2x2 maxpool.
// Grid: (256 pooled locations, 64 batch-tiles of 64). The x-patch is
// im2col'ed straight into LDS as [bl][loc][k(32)] (k=27 gets 1.0 to activate
// the bias row of wlcT) so the A fragment is two contiguous b128 LDS loads.
// 8 waves = 4 batch Mtiles x 2 output Ntiles; 4 WMMAs/wave with C=0, then
// relu∘maxpool via max-combine. Output NHWC f16 (B,16,16,32).
// ---------------------------------------------------------------------------
__global__ __launch_bounds__(256) void k_lc(
    const float* __restrict__ x, const _Float16* __restrict__ wlcT,
    _Float16* __restrict__ out1h)
{
  __shared__ __align__(16) _Float16 lds_a[64 * 4 * 32];  // [bl][loc][k] 16KB
  __shared__ __align__(16) _Float16 lds_w[4 * 32 * 32];  // [loc][o][k]  8KB
  __shared__ float lds_out[64 * 32];                     // [bl][o]      8KB

  const int tid = threadIdx.x;
  const int loc = blockIdx.x;              // i2*16 + j2
  const int bt  = blockIdx.y;
  const int i2 = loc >> 4, j2 = loc & 15;

  // im2col: element (bl, l, k) -> x[b, c, 2*i2+pi-1+dy, 2*j2+pj-1+dx]
  for (int e = tid; e < 64 * 4 * 32; e += 256) {
    int bl = e >> 7, rest = e & 127;
    int l = rest >> 5, k = rest & 31;      // k == tid&31: loop-invariant math
    float v = 0.f;
    if (k < 27) {
      int c = k / 9, dk = k - c * 9;
      int dy = dk / 3, dx = dk - dy * 3;
      int yg = 2 * i2 + (l >> 1) - 1 + dy;
      int xg = 2 * j2 + (l & 1) - 1 + dx;
      if ((unsigned)yg < 32u && (unsigned)xg < 32u)
        v = x[(((bt * 64 + bl) * 3 + c) << 10) + (yg << 5) + xg];
    } else if (k == 27) {
      v = 1.0f;                            // activates bias row of wlcT
    }
    lds_a[e] = (_Float16)v;
  }
  // weights: 4 contiguous 2KB blocks
  for (int c4 = tid; c4 < 512; c4 += 256) {
    int l = c4 >> 7, rest = c4 & 127;
    int ij = (2 * i2 + (l >> 1)) * 32 + (2 * j2 + (l & 1));
    ((int4*)lds_w)[c4] = ((const int4*)(wlcT + ij * 1024))[rest];
  }
  __syncthreads();

  const int wave = tid >> 5, lane = tid & 31;
  const int mtile = wave >> 1, ntile = wave & 1;
  const int col = lane & 15;
  const int hb = (lane & 16) ? 8 : 0;      // A K-half base
  const int kB = (lane & 16) ? 8 : 0;      // B K-half base
  const int o  = ntile * 16 + col;
  const int bl = mtile * 16 + col;         // A row = batch-local

  const v8f zc = {0.f, 0.f, 0.f, 0.f, 0.f, 0.f, 0.f, 0.f};
  v8f acc = zc;
#pragma unroll
  for (int l = 0; l < 4; ++l) {
    const _Float16* pA = &lds_a[(bl * 4 + l) * 32 + hb];
    v16h av = cat16(*(const v8h*)pA, *(const v8h*)(pA + 16));
    const _Float16* pB = &lds_w[(l * 32 + o) * 32 + kB];
    v16h bv = cat16(*(const v8h*)pB, *(const v8h*)(pB + 16));
    v8f d = WMMA_F32_F16(av, bv, zc);
#pragma unroll
    for (int r = 0; r < 8; ++r) acc[r] = fmaxf(acc[r], d[r]);  // relu∘maxpool
  }

  const int mhalf = (lane & 16) ? 8 : 0;
#pragma unroll
  for (int r = 0; r < 8; ++r)
    lds_out[(mtile * 16 + mhalf + r) * 32 + o] = acc[r];
  __syncthreads();

  const size_t obase = (size_t)bt * 64 * 8192 + (size_t)loc * 32;
  for (int e = tid; e < 2048; e += 256) {
    int bl2 = e >> 5, oo = e & 31;
    out1h[obase + (size_t)bl2 * 8192 + oo] = (_Float16)lds_out[e];
  }
}

// ---------------------------------------------------------------------------
// k_conv: 3x3 SAME conv (32->64 ch) as implicit GEMM + bias + ReLU + maxpool.
// One WG per batch. Zero-padded 18x18x32 f16 slab + [o][k] weights staged via
// async global->LDS DMA; pooled output written back via async LDS->global
// (S_ENDPGM's implicit wait-idle covers completion). 72 WMMAs/wave.
// ---------------------------------------------------------------------------
__global__ __launch_bounds__(256) void k_conv(
    const _Float16* __restrict__ in,     // (B,16,16,32) f16
    const _Float16* __restrict__ w2T,    // [64][288] f16
    const float* __restrict__ b_conv,
    _Float16* __restrict__ out2h)        // (B,64,8,8) f16, c-major
{
  __shared__ __align__(16) _Float16 lds_in[18 * 18 * 32];
  __shared__ __align__(16) _Float16 lds_w2[64 * 288];
  __shared__ __align__(16) _Float16 lds_o2[64 * 64];  // [o][yh*8+xh]

  const int tid = threadIdx.x;
  const int b = blockIdx.x;

  // phase 1: zero the padded slab; async-stage the weights
  for (int e = tid; e < 18 * 18 * 32 / 2; e += 256) ((unsigned int*)lds_in)[e] = 0u;
  for (int c4 = tid; c4 < 64 * 288 / 8; c4 += 256)
    g2l_b128(&((int4*)lds_w2)[c4], &((const int4*)w2T)[c4]);
  __syncthreads();
  // phase 2: async-stage the slab interior (disjoint from the zeroed halo)
  const int4* src = (const int4*)(in + (size_t)b * 8192);
  for (int c4 = tid; c4 < 1024; c4 += 256) {
    int p = c4 >> 2, part = c4 & 3;
    int y = p >> 4, xx = p & 15;
    g2l_b128(&((int4*)&lds_in[((y + 1) * 18 + (xx + 1)) * 32])[part], &src[c4]);
  }
  wait_async0();
  __syncthreads();

  const int wave = tid >> 5, lane = tid & 31;
  const int ntile = wave & 3, rgrp = wave >> 2;
  const int col = lane & 15;
  const int hb = (lane & 16) ? 8 : 0;
  const int kB = (lane & 16) ? 8 : 0;
  const int o = ntile * 16 + col;
  const int mrow = lane & 15;              // x coordinate
  const int mhalf = (lane & 16) ? 8 : 0;
  const float bias = b_conv[o];

  for (int ri = 0; ri < 4; ++ri) {
    const int r = rgrp * 4 + ri;           // pooled row yh
    v8f dA = {0,0,0,0,0,0,0,0}, dB = {0,0,0,0,0,0,0,0};
#pragma unroll
    for (int s = 0; s < 9; ++s) {
      const int dy = s / 3, dx = s - dy * 3;
      const _Float16* pA = &lds_in[((2 * r + dy) * 18 + (mrow + dx)) * 32 + hb];
      const _Float16* pB = &lds_in[((2 * r + 1 + dy) * 18 + (mrow + dx)) * 32 + hb];
      v16h aA = cat16(*(const v8h*)pA, *(const v8h*)(pA + 16));
      v16h aB = cat16(*(const v8h*)pB, *(const v8h*)(pB + 16));
      const _Float16* pw = &lds_w2[o * 288 + s * 32 + kB];
      v16h bv = cat16(*(const v8h*)pw, *(const v8h*)(pw + 16));
      dA = WMMA_F32_F16(aA, bv, dA);
      dB = WMMA_F32_F16(aB, bv, dB);
    }
#pragma unroll
    for (int q = 0; q < 4; ++q) {          // pool x pairs + y pair + relu
      float v = fmaxf(fmaxf(dA[2 * q], dA[2 * q + 1]),
                      fmaxf(dB[2 * q], dB[2 * q + 1]));
      v = fmaxf(v + bias, 0.f);
      lds_o2[o * 64 + r * 8 + (mhalf >> 1) + q] = (_Float16)v;
    }
  }
  __syncthreads();
  int4* dst = (int4*)(out2h + (size_t)b * 4096);
  for (int c4 = tid; c4 < 512; c4 += 256)
    l2g_b128(&dst[c4], &((const int4*)lds_o2)[c4]);
  // async stores drain before S_ENDPGM (implicit wait-idle)
}

// ---------------------------------------------------------------------------
// k_fc1: (4096x4096) @ W^T(4096x256) + bias + ReLU, f16 out. Double-buffered
// LDS ping-pong with async global->LDS DMA: stage tile kt+1 while computing
// tile kt. 64 M x 32 N per WG, K staged 256 at a time, 128 WMMAs per wave.
// ---------------------------------------------------------------------------
__global__ __launch_bounds__(256) void k_fc1(
    const _Float16* __restrict__ a,      // (4096,4096) f16
    const _Float16* __restrict__ w1h,    // [n(256)][k(4096)] f16
    const float* __restrict__ b_fc1,
    _Float16* __restrict__ outh)         // (4096,256) f16
{
  __shared__ __align__(16) _Float16 lds_a[2][64 * 256];   // [m][k], 2x32KB
  __shared__ __align__(16) _Float16 lds_b[2][32 * 256];   // [n][k], 2x16KB
  const int tid = threadIdx.x;
  const int n0 = blockIdx.x * 32;
  const int b0 = blockIdx.y * 64;

  auto stage = [&](int buf, int k0) {
    for (int c4 = tid; c4 < 2048; c4 += 256)
      g2l_b128(&((int4*)lds_a[buf])[c4],
               (const int4*)(a + (size_t)(b0 + (c4 >> 5)) * 4096 + k0) + (c4 & 31));
    for (int c4 = tid; c4 < 1024; c4 += 256)
      g2l_b128(&((int4*)lds_b[buf])[c4],
               (const int4*)(w1h + (size_t)(n0 + (c4 >> 5)) * 4096 + k0) + (c4 & 31));
  };

  const int wave = tid >> 5, lane = tid & 31;
  const int mtile = wave >> 1, ntile = wave & 1;
  const int col = lane & 15;
  const int hb = (lane & 16) ? 8 : 0;
  const int kB = (lane & 16) ? 8 : 0;
  const int mhalf = (lane & 16) ? 8 : 0;

  stage(0, 0);
  wait_async0();
  __syncthreads();

  v8f acc = {0,0,0,0,0,0,0,0};
  int cur = 0;
  for (int kt = 0; kt < 16; ++kt) {
    if (kt + 1 < 16) stage(cur ^ 1, (kt + 1) * 256);
    const _Float16* ca = lds_a[cur];
    const _Float16* cb = lds_b[cur];
#pragma unroll
    for (int ks = 0; ks < 8; ++ks) {
      const _Float16* pA = &ca[(mtile * 16 + col) * 256 + ks * 32 + hb];
      v16h av = cat16(*(const v8h*)pA, *(const v8h*)(pA + 16));
      const _Float16* pB = &cb[(ntile * 16 + col) * 256 + ks * 32 + kB];
      v16h bv = cat16(*(const v8h*)pB, *(const v8h*)(pB + 16));
      acc = WMMA_F32_F16(av, bv, acc);
    }
    wait_async0();
    __syncthreads();
    cur ^= 1;
  }
  const int n = n0 + ntile * 16 + col;
  const float bias = b_fc1[n];
#pragma unroll
  for (int r = 0; r < 8; ++r) {
    int bb = b0 + mtile * 16 + mhalf + r;
    float v = fmaxf(acc[r] + bias, 0.f);
    outh[(size_t)bb * 256 + n] = (_Float16)v;
  }
}

// ---------------------------------------------------------------------------
// k_fc2: (4096x256) @ W^T(256x16pad) + bias -> fp32 logits (N=10 valid).
// Both operands straight from global (16B-aligned contiguous fragments);
// store masks lanes with col>=10 (after the WMMA, EXEC all-1 during it).
// ---------------------------------------------------------------------------
__global__ __launch_bounds__(256) void k_fc2(
    const _Float16* __restrict__ a,      // (4096,256) f16
    const _Float16* __restrict__ w3T,    // [16][256] f16
    const float* __restrict__ b_fc2,
    float* __restrict__ out)             // (4096,10) f32
{
  const int tid = threadIdx.x;
  const int wave = tid >> 5, lane = tid & 31;
  const int gw = blockIdx.x * 8 + wave;  // 0..255 -> 16 batches each
  const int b0 = gw * 16;
  const int col = lane & 15;
  const int hb = (lane & 16) ? 8 : 0;
  const int kB = (lane & 16) ? 8 : 0;
  const int mhalf = (lane & 16) ? 8 : 0;

  v8f acc = {0,0,0,0,0,0,0,0};
#pragma unroll
  for (int ks = 0; ks < 8; ++ks) {
    const _Float16* pA = a + (size_t)(b0 + col) * 256 + ks * 32 + hb;
    v16h av = cat16(*(const v8h*)pA, *(const v8h*)(pA + 16));
    const _Float16* pB = w3T + col * 256 + ks * 32 + kB;
    v16h bv = cat16(*(const v8h*)pB, *(const v8h*)(pB + 16));
    acc = WMMA_F32_F16(av, bv, acc);
  }
  if (col < 10) {
    const float bias = b_fc2[col];
#pragma unroll
    for (int r = 0; r < 8; ++r)
      out[(size_t)(b0 + mhalf + r) * 10 + col] = acc[r] + bias;
  }
}

// ---------------------------------------------------------------------------
extern "C" void kernel_launch(void* const* d_in, const int* in_sizes, int n_in,
                              void* d_out, int out_size, void* d_ws, size_t ws_size,
                              hipStream_t stream) {
  (void)in_sizes; (void)n_in; (void)out_size; (void)ws_size;
  const float* x      = (const float*)d_in[0];
  const float* w_lc   = (const float*)d_in[1];
  const float* b_lc   = (const float*)d_in[2];
  const float* w_conv = (const float*)d_in[3];
  const float* b_conv = (const float*)d_in[4];
  const float* w_fc1  = (const float*)d_in[5];
  const float* b_fc1  = (const float*)d_in[6];
  const float* w_fc2  = (const float*)d_in[7];
  const float* b_fc2  = (const float*)d_in[8];

  // Workspace layout (bytes; all offsets 16B-aligned). Total ~107 MB.
  char* ws = (char*)d_ws;
  _Float16* wlcT  = (_Float16*)(ws + 0);          //  2,097,152  [ij][32][32]
  _Float16* w1h   = (_Float16*)(ws + 2097152);    //  2,097,152  [256][4096]
  _Float16* w2T   = (_Float16*)(ws + 4194304);    //     36,864  [64][288]
  _Float16* w3T   = (_Float16*)(ws + 4231168);    //      8,192  [16][256]
  _Float16* out1h = (_Float16*)(ws + 4239360);    // 67,108,864  (B,16,16,32)
  _Float16* out2h = (_Float16*)(ws + 71348224);   // 33,554,432  (B,4096)
  _Float16* fc1h  = (_Float16*)(ws + 104902656);  //  2,097,152  (B,256)

  k_prep<<<(2119680 + 255) / 256, 256, 0, stream>>>(
      w_lc, b_lc, w_conv, w_fc1, w_fc2, wlcT, w1h, w2T, w3T);
  k_lc  <<<dim3(256, 64), 256, 0, stream>>>(x, wlcT, out1h);
  k_conv<<<4096, 256, 0, stream>>>(out1h, w2T, b_conv, out2h);
  k_fc1 <<<dim3(8, 64), 256, 0, stream>>>(out2h, w1h, b_fc1, fc1h);
  k_fc2 <<<32, 256, 0, stream>>>(fc1h, w3T, b_fc2, (float*)d_out);
}
